// NeuralODE_66331474919772
// MI455X (gfx1250) — matrix operand order repaired and checked by reading, not compiled, not run on hardware
//
#include <hip/hip_runtime.h>
#include <hip/hip_bf16.h>

typedef __attribute__((ext_vector_type(16))) __bf16 bf16x16;
typedef __attribute__((ext_vector_type(8)))  __bf16 bf16x8;
typedef __attribute__((ext_vector_type(8)))  float  f32x8;

#define Bsz 4096
#define Dd  256
#define Ww  1024
#define Tt  33
#define MT  32              // batch rows per workgroup
#define NBLK (Bsz / MT)     // 128 workgroups
#define NTHR 256            // 8 waves (wave32)
#define PAD 8               // LDS row padding in bf16 elems (16 B) -> kills bank conflicts

// ---- workspace layout (bytes) ----
#define OFF_W1B 0
#define OFF_W2B (OFF_W1B + (size_t)Ww * Dd * 2)
#define OFF_W3B (OFF_W2B + (size_t)Ww * Ww * 2)
#define OFF_W4B (OFF_W3B + (size_t)Ww * Ww * 2)
#define OFF_Y   (OFF_W4B + (size_t)Dd * Ww * 2)
#define OFF_K   (OFF_Y + (size_t)Bsz * Dd * 4)
// OFF_K + 6*Bsz*Dd*4  ~= 34.6 MB total scratch

// ---- LDS layout (bytes) ----
#define LDS_A0   0
#define LDS_BUFA (MT * (Dd + PAD) * 2)                       // 16896
#define LDS_BUFB (LDS_BUFA + MT * (Ww + PAD) * 2)            // 82944
#define LDS_TOT  (LDS_BUFB + MT * (Ww + PAD) * 2)            // 148992 (2 WGs / 320KB WGP)

// Dopri5 tableau (autonomous field, fixed step)
__constant__ float c_AT[6][5] = {
    {0.f, 0.f, 0.f, 0.f, 0.f},
    {1.f / 5.f, 0.f, 0.f, 0.f, 0.f},
    {3.f / 40.f, 9.f / 40.f, 0.f, 0.f, 0.f},
    {44.f / 45.f, -56.f / 15.f, 32.f / 9.f, 0.f, 0.f},
    {19372.f / 6561.f, -25360.f / 2187.f, 64448.f / 6561.f, -212.f / 729.f, 0.f},
    {9017.f / 3168.f, -355.f / 33.f, 46732.f / 5247.f, 49.f / 176.f, -5103.f / 18656.f},
};
#define CB1 (35.f / 384.f)
#define CB3 (500.f / 1113.f)
#define CB4 (125.f / 192.f)
#define CB5 (-2187.f / 6784.f)
#define CB6 (11.f / 84.f)

__device__ __forceinline__ __bf16 to_bf16(float f) { return (__bf16)f; }

// GEMM over one 32-row batch tile: Out[MT x N] = act(Ain[MT x K] @ Wb^T + bias)
// Ain: LDS bf16, row stride K+PAD.  Wb: global bf16 row-major N x K (dense, no padding).
// Each of the 8 waves owns a disjoint N/8 column range and the FULL 32 rows, so one
// B fragment is reused by two WMMAs (rows 0-15 and 16-31): 1 global b128 per WMMA.
template <int K, int N, bool RELU, bool TO_LDS>
__device__ __forceinline__ void gemm_tile(const __bf16* __restrict__ Ain,
                                          __bf16* __restrict__ outL,
                                          float* __restrict__ outG,  // tile base, row stride Dd
                                          const __bf16* __restrict__ Wb,
                                          const float* __restrict__ bias,
                                          int lane, int wave) {
    constexpr int LDA   = K + PAD;
    constexpr int LDO   = N + PAD;
    constexpr int WCOLS = N / 8;                       // columns per wave
    constexpr int SW    = (WCOLS >= 64) ? 64 : WCOLS;  // strip width
    constexpr int NT    = SW / 16;                     // 16-col acc tiles per strip
    const int l15 = lane & 15;
    const int lhi = lane >> 4;  // which K-half this lane carries
    // A frag rows (canonical 16-bit A layout: lanes 0-15 K{0..7,16..23}, 16-31 K{8..15,24..31})
    const __bf16* arow0 = Ain + l15 * LDA + 8 * lhi;         // rows 0..15
    const __bf16* arow1 = Ain + (16 + l15) * LDA + 8 * lhi;  // rows 16..31

    for (int n0 = wave * WCOLS; n0 < wave * WCOLS + WCOLS; n0 += SW) {
        f32x8 acc0[NT], acc1[NT];
#pragma unroll
        for (int j = 0; j < NT; ++j) {
            const float bv = bias[n0 + 16 * j + l15];  // C layout: col = lane&15 for all slots
#pragma unroll
            for (int v = 0; v < 8; ++v) { acc0[j][v] = bv; acc1[j][v] = bv; }
        }
#pragma unroll 2
        for (int k0 = 0; k0 < K; k0 += 32) {
            const bf16x8 a0lo = *(const bf16x8*)(arow0 + k0);
            const bf16x8 a0hi = *(const bf16x8*)(arow0 + k0 + 16);
            const bf16x8 a1lo = *(const bf16x8*)(arow1 + k0);
            const bf16x8 a1hi = *(const bf16x8*)(arow1 + k0 + 16);
            bf16x16 af0, af1;
#pragma unroll
            for (int i = 0; i < 8; ++i) {
                af0[i] = a0lo[i]; af0[i + 8] = a0hi[i];
                af1[i] = a1lo[i]; af1[i + 8] = a1hi[i];
            }
#pragma unroll
            for (int j = 0; j < NT; ++j) {
                // B 32x16: lane = column n, K{8*lhi..+7, 16+8*lhi..+7} (matches A layout)
                const __bf16* bp = Wb + (size_t)(n0 + 16 * j + l15) * K + k0 + 8 * lhi;
                const bf16x8 blo = *(const bf16x8*)(bp);
                const bf16x8 bhi = *(const bf16x8*)(bp + 16);
                bf16x16 bf;
#pragma unroll
                for (int i = 0; i < 8; ++i) { bf[i] = blo[i]; bf[i + 8] = bhi[i]; }
                acc0[j] = __builtin_amdgcn_wmma_f32_16x16x32_bf16(false, af0, false, bf, (short)0,
                                                                  acc0[j], false, false);
                acc1[j] = __builtin_amdgcn_wmma_f32_16x16x32_bf16(false, af1, false, bf, (short)0,
                                                                  acc1[j], false, false);
            }
        }
        // Epilogue: C/D layout row m = v + 8*lhi (+16 for acc1), col n = n0+16j+(lane&15)
#pragma unroll
        for (int j = 0; j < NT; ++j) {
            const int n = n0 + 16 * j + l15;
#pragma unroll
            for (int v = 0; v < 8; ++v) {
                const int m0 = v + 8 * lhi;
                float v0 = acc0[j][v];
                float v1 = acc1[j][v];
                if (RELU) { v0 = fmaxf(v0, 0.0f); v1 = fmaxf(v1, 0.0f); }
                if (TO_LDS) {
                    outL[m0 * LDO + n]        = to_bf16(v0);
                    outL[(m0 + 16) * LDO + n] = to_bf16(v1);
                } else {
                    outG[(size_t)m0 * Dd + n]        = v0;
                    outG[(size_t)(m0 + 16) * Dd + n] = v1;
                }
            }
        }
    }
}

__global__ void __launch_bounds__(NTHR)
ode_prep(const float* __restrict__ W1, const float* __restrict__ W2,
         const float* __restrict__ W3, const float* __restrict__ W4,
         const float* __restrict__ y0, __bf16* __restrict__ W1b, __bf16* __restrict__ W2b,
         __bf16* __restrict__ W3b, __bf16* __restrict__ W4b, float* __restrict__ y_cur,
         float* __restrict__ out) {
    const size_t stride = (size_t)gridDim.x * blockDim.x;
    for (size_t i = (size_t)blockIdx.x * blockDim.x + threadIdx.x; i < (size_t)Ww * Ww;
         i += stride) {
        if (i < (size_t)Ww * Dd) { W1b[i] = to_bf16(W1[i]); W4b[i] = to_bf16(W4[i]); }
        W2b[i] = to_bf16(W2[i]);
        W3b[i] = to_bf16(W3[i]);
        if (i < (size_t)Bsz * Dd) {
            const float v = y0[i];
            y_cur[i] = v;   // integration state
            out[i]   = v;   // ys[0] = y0
        }
    }
}

__global__ void __launch_bounds__(NTHR)
ode_main(const float* __restrict__ ts, const float* __restrict__ b1,
         const float* __restrict__ b2, const float* __restrict__ b3,
         const float* __restrict__ b4, const __bf16* __restrict__ W1b,
         const __bf16* __restrict__ W2b, const __bf16* __restrict__ W3b,
         const __bf16* __restrict__ W4b, float* __restrict__ y_cur,
         float* __restrict__ kbuf, float* __restrict__ out) {
    extern __shared__ char smem[];
    __bf16* a0   = (__bf16*)(smem + LDS_A0);    // MT x (256+PAD) bf16
    __bf16* bufA = (__bf16*)(smem + LDS_BUFA);  // MT x (1024+PAD) bf16
    __bf16* bufB = (__bf16*)(smem + LDS_BUFB);  // MT x (1024+PAD) bf16

    const int tid   = threadIdx.x;
    const int lane  = tid & 31;
    const int wave  = tid >> 5;
    const int row0  = blockIdx.x * MT;
    const size_t KS = (size_t)Bsz * Dd;

    for (int t = 0; t < Tt - 1; ++t) {
        const float dt = ts[t + 1] - ts[t];
        for (int s = 0; s < 6; ++s) {
            // Stage input: a0 = bf16( y + dt * sum_j A[s][j] * k_j ), elementwise
            for (int i = tid; i < MT * Dd; i += NTHR) {
                const int r = i >> 8, c = i & 255;
                const size_t g = (size_t)(row0 + r) * Dd + c;
                float v = y_cur[g];
                for (int j = 0; j < s; ++j) v += dt * c_AT[s][j] * kbuf[(size_t)j * KS + g];
                a0[r * (Dd + PAD) + c] = to_bf16(v);
            }
            __syncthreads();
            gemm_tile<Dd, Ww, true, true>(a0, bufA, nullptr, W1b, b1, lane, wave);
            __syncthreads();
            gemm_tile<Ww, Ww, true, true>(bufA, bufB, nullptr, W2b, b2, lane, wave);
            __syncthreads();
            gemm_tile<Ww, Ww, true, true>(bufB, bufA, nullptr, W3b, b3, lane, wave);
            __syncthreads();
            gemm_tile<Ww, Dd, false, false>(bufA, nullptr,
                                            kbuf + (size_t)s * KS + (size_t)row0 * Dd, W4b, b4,
                                            lane, wave);
            __threadfence();  // k_s written by all waves must be visible to whole workgroup
            __syncthreads();
        }
        // Combine: y += dt*(B1 k1 + B3 k3 + B4 k4 + B5 k5 + B6 k6); emit ys[t+1]
        for (int i = tid; i < MT * Dd; i += NTHR) {
            const size_t g = (size_t)(row0 + (i >> 8)) * Dd + (i & 255);
            float y = y_cur[g];
            y += dt * (CB1 * kbuf[g] + CB3 * kbuf[2 * KS + g] + CB4 * kbuf[3 * KS + g] +
                       CB5 * kbuf[4 * KS + g] + CB6 * kbuf[5 * KS + g]);
            y_cur[g]                      = y;
            out[(size_t)(t + 1) * KS + g] = y;
        }
        __syncthreads();
    }
}

extern "C" void kernel_launch(void* const* d_in, const int* in_sizes, int n_in, void* d_out,
                              int out_size, void* d_ws, size_t ws_size, hipStream_t stream) {
    const float* y0 = (const float*)d_in[0];
    const float* ts = (const float*)d_in[1];
    const float* W1 = (const float*)d_in[2];
    const float* b1 = (const float*)d_in[3];
    const float* W2 = (const float*)d_in[4];
    const float* b2 = (const float*)d_in[5];
    const float* W3 = (const float*)d_in[6];
    const float* b3 = (const float*)d_in[7];
    const float* W4 = (const float*)d_in[8];
    const float* b4 = (const float*)d_in[9];
    float* out = (float*)d_out;

    char* ws     = (char*)d_ws;
    __bf16* W1b  = (__bf16*)(ws + OFF_W1B);
    __bf16* W2b  = (__bf16*)(ws + OFF_W2B);
    __bf16* W3b  = (__bf16*)(ws + OFF_W3B);
    __bf16* W4b  = (__bf16*)(ws + OFF_W4B);
    float* y_cur = (float*)(ws + OFF_Y);
    float* kbuf  = (float*)(ws + OFF_K);

    (void)hipFuncSetAttribute((const void*)ode_main, hipFuncAttributeMaxDynamicSharedMemorySize,
                              LDS_TOT);

    ode_prep<<<2048, NTHR, 0, stream>>>(W1, W2, W3, W4, y0, W1b, W2b, W3b, W4b, y_cur, out);
    ode_main<<<NBLK, NTHR, LDS_TOT, stream>>>(ts, b1, b2, b3, b4, W1b, W2b, W3b, W4b, y_cur,
                                              kbuf, out);
}
